// ParentGRU_34900904247524
// MI455X (gfx1250) — matrix operand order repaired and checked by reading, not compiled, not run on hardware
//
#include <hip/hip_runtime.h>
#include <hip/hip_bf16.h>

typedef __attribute__((ext_vector_type(16))) __bf16 v16bf;
typedef __attribute__((ext_vector_type(8)))  __bf16 v8bf;
typedef __attribute__((ext_vector_type(8)))  float  v8f;
typedef __attribute__((ext_vector_type(4)))  float  v4f;
typedef __attribute__((ext_vector_type(4)))  unsigned short us4;

#define IN_SIZE 512
#define HID     512
#define BATCH   65536
#define MTILE   64                 // rows per block (4 WMMA sub-tiles of 16)
#define MSUB    4
#define WMAT    (HID * IN_SIZE)    // 262144 elements per weight matrix

union Frag { v16bf v; v8bf h[2]; };

__device__ __forceinline__ unsigned short f2bf(float f) {
  unsigned u = __float_as_uint(f);
  u += 0x7FFFu + ((u >> 16) & 1u);   // round-to-nearest-even
  return (unsigned short)(u >> 16);
}
__device__ __forceinline__ float bf2f(unsigned short h) {
  return __uint_as_float(((unsigned)h) << 16);
}
__device__ __forceinline__ v8f wmma_bf16(v16bf a, v16bf b, v8f c) {
  return __builtin_amdgcn_wmma_f32_16x16x32_bf16(
      /*neg_a=*/false, a, /*neg_b=*/false, b,
      /*c_mod=*/(short)0, c, /*reuse_a=*/false, /*reuse_b=*/false);
}
__device__ __forceinline__ float sigmoidf_fast(float x) {
  return 1.0f / (1.0f + __expf(-x));
}

// ---------------- weight conversion pre-pass: f32 -> bf16 into d_ws ----------
struct WPtrs { const float* p[12]; };

__global__ void cvt_weights(WPtrs wp, unsigned short* __restrict__ dst) {
  const int mat = blockIdx.y;
  const int idx = (blockIdx.x * blockDim.x + threadIdx.x) * 4;
  const float* src = wp.p[mat];
  v4f v = *(const v4f*)(src + idx);
  us4 o;
  o.x = f2bf(v.x); o.y = f2bf(v.y); o.z = f2bf(v.z); o.w = f2bf(v.w);
  *(us4*)(dst + (size_t)mat * WMAT + idx) = o;
}

// ---------------- fused ParentGRU cell --------------------------------------
__global__ __launch_bounds__(256, 1)
void parent_gru_kernel(const float* __restrict__ x,  const float* __restrict__ st,
                       const float* __restrict__ pa,
                       const float* __restrict__ ba, const float* __restrict__ br,
                       const float* __restrict__ br1, const float* __restrict__ bc,
                       const unsigned short* __restrict__ wbf,
                       float* __restrict__ out) {
  // dynamic LDS: 5 x (64 x 512) bf16 = 320 KB (ISA max per workgroup)
  extern __shared__ __align__(16) unsigned short smem[];
  unsigned short* sIn = smem;                  // inputs (original, kept)
  unsigned short* sSt = smem + 1 * MTILE * 512;  // state (original, kept)
  unsigned short* sPa = smem + 2 * MTILE * 512;  // parent (original, kept)
  unsigned short* sRs = smem + 3 * MTILE * 512;  // sigmoid(r)  * state
  unsigned short* sR1 = smem + 4 * MTILE * 512;  // sigmoid(r1) * parent

  const int tid  = threadIdx.x;
  const int lane = tid & 31;
  const int wave = tid >> 5;
  const int row0 = blockIdx.x * MTILE;

  // ---- stage activation tiles to LDS as bf16 (64 rows x 512, contiguous) ---
  {
    const float* srcs[3]    = {x, st, pa};
    unsigned short* dsts[3] = {sIn, sSt, sPa};
    for (int m3 = 0; m3 < 3; ++m3) {
      const float* s = srcs[m3] + (size_t)row0 * 512;
      unsigned short* d = dsts[m3];
#pragma unroll 4
      for (int ii = 0; ii < 32; ++ii) {
        int off = (tid + 256 * ii) * 4;          // 0..32767
        v4f v = __builtin_nontemporal_load((const v4f*)(s + off));
        us4 o;
        o.x = f2bf(v.x); o.y = f2bf(v.y); o.z = f2bf(v.z); o.w = f2bf(v.w);
        *(us4*)(d + off) = o;
      }
    }
  }
  __syncthreads();

  // weight matrix bases in ws (bf16, same [out,in] row-major layout as torch)
  const unsigned short* Wa  = wbf + 0  * (size_t)WMAT;
  const unsigned short* Wr  = wbf + 1  * (size_t)WMAT;
  const unsigned short* Wr1 = wbf + 2  * (size_t)WMAT;
  const unsigned short* Wc  = wbf + 3  * (size_t)WMAT;
  const unsigned short* Ua  = wbf + 4  * (size_t)WMAT;
  const unsigned short* Ur  = wbf + 5  * (size_t)WMAT;
  const unsigned short* Ur1 = wbf + 6  * (size_t)WMAT;
  const unsigned short* Uc  = wbf + 7  * (size_t)WMAT;
  const unsigned short* Pa  = wbf + 8  * (size_t)WMAT;
  const unsigned short* Pr  = wbf + 9  * (size_t)WMAT;
  const unsigned short* Pr1 = wbf + 10 * (size_t)WMAT;
  const unsigned short* Pc  = wbf + 11 * (size_t)WMAT;

  // fragment lane geometry (ISA 7.12.2 layouts)
  const int mA    = lane & 15;            // A row within sub-tile
  const int koffA = (lane >> 4) * 8;      // A: 8-element K chunks per lane half
  const int colB  = lane & 15;            // B/C/D column within tile
  const int koffB = (lane >> 4) * 8;      // B: 8-element K chunks per lane half
  const int mrow  = (lane >> 4) * 8;      // C/D: M base per lane half

  const v8f zero = {0.f, 0.f, 0.f, 0.f, 0.f, 0.f, 0.f, 0.f};

#define LDB(MAT) do { const unsigned short* pb_ = (MAT) + wo;                   \
                      b.h[0] = *(const v8bf*)pb_; b.h[1] = *(const v8bf*)(pb_ + 16); } while (0)
#define LDA(DST, SRC) do {                                                      \
    const unsigned short* pa_ = (SRC) + (s * 16 + mA) * 512 + k0 + koffA;       \
    (DST).h[0] = *(const v8bf*)pa_; (DST).h[1] = *(const v8bf*)(pa_ + 16); } while (0)

  // ================= pass R: r and r1 pre-activations =======================
  v8f acc_r[MSUB][4], acc_r1[MSUB][4];
#pragma unroll
  for (int s = 0; s < MSUB; ++s)
#pragma unroll
    for (int i = 0; i < 4; ++i) { acc_r[s][i] = zero; acc_r1[s][i] = zero; }

#pragma unroll 1
  for (int k0 = 0; k0 < IN_SIZE; k0 += 32) {
    Frag aIn[MSUB], aSt[MSUB], aPa[MSUB];
#pragma unroll
    for (int s = 0; s < MSUB; ++s) { LDA(aIn[s], sIn); LDA(aSt[s], sSt); LDA(aPa[s], sPa); }
#pragma unroll
    for (int i = 0; i < 4; ++i) {
      const int n = (wave * 4 + i) * 16 + colB;
      const size_t wo = (size_t)n * 512 + k0 + koffB;
      Frag b;
      LDB(Wr);
#pragma unroll
      for (int s = 0; s < MSUB; ++s) acc_r[s][i]  = wmma_bf16(aIn[s].v, b.v, acc_r[s][i]);
      LDB(Ur);
#pragma unroll
      for (int s = 0; s < MSUB; ++s) acc_r[s][i]  = wmma_bf16(aSt[s].v, b.v, acc_r[s][i]);
      LDB(Pr);
#pragma unroll
      for (int s = 0; s < MSUB; ++s) acc_r[s][i]  = wmma_bf16(aPa[s].v, b.v, acc_r[s][i]);
      LDB(Wr1);
#pragma unroll
      for (int s = 0; s < MSUB; ++s) acc_r1[s][i] = wmma_bf16(aIn[s].v, b.v, acc_r1[s][i]);
      LDB(Ur1);
#pragma unroll
      for (int s = 0; s < MSUB; ++s) acc_r1[s][i] = wmma_bf16(aSt[s].v, b.v, acc_r1[s][i]);
      LDB(Pr1);
#pragma unroll
      for (int s = 0; s < MSUB; ++s) acc_r1[s][i] = wmma_bf16(aPa[s].v, b.v, acc_r1[s][i]);
    }
  }

  // ---- elementwise: rs = sigmoid(pre_r)*state -> sRs ; r1p -> sR1 ----------
  // (fresh LDS regions; originals untouched. each element owned by one thread)
#pragma unroll
  for (int s = 0; s < MSUB; ++s)
#pragma unroll
    for (int i = 0; i < 4; ++i) {
      const int n = (wave * 4 + i) * 16 + colB;
      const float biasr  = br[n];
      const float biasr1 = br1[n];
#pragma unroll
      for (int j = 0; j < 8; ++j) {
        const int m = s * 16 + mrow + j;
        const float rv  = sigmoidf_fast(acc_r[s][i][j]  + biasr);
        const float r1v = sigmoidf_fast(acc_r1[s][i][j] + biasr1);
        sRs[m * 512 + n] = f2bf(rv  * bf2f(sSt[m * 512 + n]));
        sR1[m * 512 + n] = f2bf(r1v * bf2f(sPa[m * 512 + n]));
      }
    }
  // acc_r / acc_r1 dead here

  // ================= pass AC: att pre-activation + input path of c ==========
  v8f acc_a[MSUB][4], acc_c[MSUB][4];
#pragma unroll
  for (int s = 0; s < MSUB; ++s)
#pragma unroll
    for (int i = 0; i < 4; ++i) { acc_a[s][i] = zero; acc_c[s][i] = zero; }

#pragma unroll 1
  for (int k0 = 0; k0 < IN_SIZE; k0 += 32) {
    Frag aIn[MSUB], aSt[MSUB], aPa[MSUB];
#pragma unroll
    for (int s = 0; s < MSUB; ++s) { LDA(aIn[s], sIn); LDA(aSt[s], sSt); LDA(aPa[s], sPa); }
#pragma unroll
    for (int i = 0; i < 4; ++i) {
      const int n = (wave * 4 + i) * 16 + colB;
      const size_t wo = (size_t)n * 512 + k0 + koffB;
      Frag b;
      LDB(Wa);
#pragma unroll
      for (int s = 0; s < MSUB; ++s) acc_a[s][i] = wmma_bf16(aIn[s].v, b.v, acc_a[s][i]);
      LDB(Ua);
#pragma unroll
      for (int s = 0; s < MSUB; ++s) acc_a[s][i] = wmma_bf16(aSt[s].v, b.v, acc_a[s][i]);
      LDB(Pa);
#pragma unroll
      for (int s = 0; s < MSUB; ++s) acc_a[s][i] = wmma_bf16(aPa[s].v, b.v, acc_a[s][i]);
      LDB(Wc);
#pragma unroll
      for (int s = 0; s < MSUB; ++s) acc_c[s][i] = wmma_bf16(aIn[s].v, b.v, acc_c[s][i]);
    }
  }
  __syncthreads();   // all rs/r1p written before any thread reads them

  // ================= pass 3: c += rs @ Uc.T + r1p @ Pc.T ====================
#pragma unroll 1
  for (int k0 = 0; k0 < HID; k0 += 32) {
    Frag aRs[MSUB], aR1[MSUB];
#pragma unroll
    for (int s = 0; s < MSUB; ++s) { LDA(aRs[s], sRs); LDA(aR1[s], sR1); }
#pragma unroll
    for (int i = 0; i < 4; ++i) {
      const int n = (wave * 4 + i) * 16 + colB;
      const size_t wo = (size_t)n * 512 + k0 + koffB;
      Frag b;
      LDB(Uc);
#pragma unroll
      for (int s = 0; s < MSUB; ++s) acc_c[s][i] = wmma_bf16(aRs[s].v, b.v, acc_c[s][i]);
      LDB(Pc);
#pragma unroll
      for (int s = 0; s < MSUB; ++s) acc_c[s][i] = wmma_bf16(aR1[s].v, b.v, acc_c[s][i]);
    }
  }

  // ---- epilogue: new_h = att*c + (1-att)*state (f32 state from global) -----
#pragma unroll
  for (int s = 0; s < MSUB; ++s)
#pragma unroll
    for (int i = 0; i < 4; ++i) {
      const int n = (wave * 4 + i) * 16 + colB;
      const float biasa = ba[n];
      const float biasc = bc[n];
#pragma unroll
      for (int j = 0; j < 8; ++j) {
        const int m = s * 16 + mrow + j;
        const float att = sigmoidf_fast(acc_a[s][i][j] + biasa);
        const float cv  = tanhf(acc_c[s][i][j] + biasc);
        const size_t gi = (size_t)(row0 + m) * 512 + n;
        const float sv = __builtin_nontemporal_load(st + gi);
        __builtin_nontemporal_store(att * cv + (1.0f - att) * sv, out + gi);
      }
    }
#undef LDB
#undef LDA
}

extern "C" void kernel_launch(void* const* d_in, const int* in_sizes, int n_in,
                              void* d_out, int out_size, void* d_ws, size_t ws_size,
                              hipStream_t stream) {
  const float* x  = (const float*)d_in[0];
  const float* st = (const float*)d_in[1];
  const float* pa = (const float*)d_in[2];
  const float* ba  = (const float*)d_in[4];
  const float* br  = (const float*)d_in[6];
  const float* br1 = (const float*)d_in[8];
  const float* bc  = (const float*)d_in[10];
  float* out = (float*)d_out;
  unsigned short* wbf = (unsigned short*)d_ws;   // 12 * 512*512 bf16 = 6 MB

  // weight order in ws: Wa Wr Wr1 Wc Ua Ur Ur1 Uc Pa Pr Pr1 Pc
  WPtrs wp;
  const int widx[12] = {3, 5, 7, 9, 11, 12, 13, 14, 15, 16, 17, 18};
  for (int i = 0; i < 12; ++i) wp.p[i] = (const float*)d_in[widx[i]];

  dim3 gcvt(WMAT / (256 * 4), 12);               // 256 blocks x 12 matrices
  cvt_weights<<<gcvt, 256, 0, stream>>>(wp, wbf);

  const size_t lds_bytes = 5u * MTILE * 512u * sizeof(unsigned short);  // 320 KB
  parent_gru_kernel<<<BATCH / MTILE, 256, lds_bytes, stream>>>(
      x, st, pa, ba, br, br1, bc, wbf, out);
}
